// MoD_18056042512561
// MI455X (gfx1250) — compile-verified, matrix operand order checked by reading, not run on hardware
//
#include <hip/hip_runtime.h>
#include <hip/hip_bf16.h>
#include <math.h>

// ---------------- problem constants ----------------
#define BATCH   4
#define SEQ     4096
#define DMODEL  2048
#define HID     8192
#define KSEL    491          // int(SEQ * 0.12)
#define KPAD    512          // padded capacity (multiple of 128)

// ---------------- types ----------------
typedef __bf16 bf16x16 __attribute__((ext_vector_type(16)));
typedef float  f32x8   __attribute__((ext_vector_type(8)));

union Frag16 { bf16x16 v; unsigned u[8]; };

// fp32 -> bf16 round-to-nearest-even
__device__ __forceinline__ unsigned short f2bf(float f) {
    unsigned u = __float_as_uint(f);
    u += 0x7FFFu + ((u >> 16) & 1u);
    return (unsigned short)(u >> 16);
}
__device__ __forceinline__ unsigned pack2bf(float lo, float hi) {
    return (unsigned)f2bf(lo) | ((unsigned)f2bf(hi) << 16);
}

// =====================================================================
// 1) Router: logits = x.Wr, auxlog = x.Wa  (one wave32 per token)
// =====================================================================
__global__ __launch_bounds__(256)
void router_kernel(const float* __restrict__ x,
                   const float* __restrict__ Wr,
                   const float* __restrict__ Wa,
                   float* __restrict__ logits,
                   float* __restrict__ auxlog,
                   float* __restrict__ aux_slot) {
    if (blockIdx.x == 0 && threadIdx.x == 0) *aux_slot = 0.0f;
    int wave = threadIdx.x >> 5;
    int lane = threadIdx.x & 31;
    int token = blockIdx.x * 8 + wave;               // 8 waves per block, grid exact
    const float4* xr  = (const float4*)(x + (size_t)token * DMODEL);
    const float4* wr4 = (const float4*)Wr;
    const float4* wa4 = (const float4*)Wa;
    float dr = 0.f, da = 0.f;
#pragma unroll
    for (int i = 0; i < 16; ++i) {
        int p = i * 32 + lane;                       // 512 float4 per row
        float4 xv = xr[p];
        float4 a  = wr4[p];
        float4 c  = wa4[p];
        dr += xv.x * a.x + xv.y * a.y + xv.z * a.z + xv.w * a.w;
        da += xv.x * c.x + xv.y * c.y + xv.z * c.z + xv.w * c.w;
    }
#pragma unroll
    for (int off = 16; off > 0; off >>= 1) {
        dr += __shfl_xor(dr, off, 32);
        da += __shfl_xor(da, off, 32);
    }
    if (lane == 0) { logits[token] = dr; auxlog[token] = da; }
}

// =====================================================================
// 2) Top-k per batch: bitonic sort 4096 (value desc, index asc tiebreak),
//    take first 491, re-sort those by index ascending. One WG per batch.
// =====================================================================
__global__ __launch_bounds__(512)
void topk_kernel(const float* __restrict__ logits,
                 int* __restrict__ sel,
                 float* __restrict__ rw) {
    __shared__ float sv[SEQ];
    __shared__ int   si[SEQ];
    const int b = blockIdx.x;
    const int tid = threadIdx.x;

    for (int i = tid; i < SEQ; i += 512) { sv[i] = logits[b * SEQ + i]; si[i] = i; }
    __syncthreads();

    // sort so that "better" (larger value, lower index on tie) comes first
    for (int k2 = 2; k2 <= SEQ; k2 <<= 1) {
        for (int j = k2 >> 1; j > 0; j >>= 1) {
            for (int i = tid; i < SEQ; i += 512) {
                int ixj = i ^ j;
                if (ixj > i) {
                    float v1 = sv[i], v2 = sv[ixj];
                    int   i1 = si[i], i2 = si[ixj];
                    bool better2 = (v2 > v1) || (v2 == v1 && i2 < i1); // ixj better
                    bool up = ((i & k2) == 0);
                    if (up ? better2 : !better2) {
                        sv[i] = v2; sv[ixj] = v1;
                        si[i] = i2; si[ixj] = i1;
                    }
                }
            }
            __syncthreads();
        }
    }

    // pad 491..511 then sort first 512 by index ascending
    for (int i = tid; i < KPAD; i += 512)
        if (i >= KSEL) { si[i] = 0x7FFFFFFF; sv[i] = 0.f; }
    __syncthreads();
    for (int k2 = 2; k2 <= KPAD; k2 <<= 1) {
        for (int j = k2 >> 1; j > 0; j >>= 1) {
            int i = tid;
            if (i < KPAD) {
                int ixj = i ^ j;
                if (ixj > i && ixj < KPAD) {
                    bool less2 = si[ixj] < si[i];
                    bool up = ((i & k2) == 0);
                    if (up ? less2 : !less2) {
                        float tv = sv[i]; sv[i] = sv[ixj]; sv[ixj] = tv;
                        int   ti = si[i]; si[i] = si[ixj]; si[ixj] = ti;
                    }
                }
            }
            __syncthreads();
        }
    }
    if (tid < KPAD) {
        bool real = tid < KSEL;
        sel[b * KPAD + tid] = real ? si[tid] : -1;
        rw [b * KPAD + tid] = real ? sv[tid] : 0.f;
    }
}

// =====================================================================
// 3a) zero selection flags
// =====================================================================
__global__ void zero_flags_kernel(int* __restrict__ flags) {
    flags[blockIdx.x * blockDim.x + threadIdx.x] = 0;  // grid exact: B*S
}

// 3b) gather selected tokens -> bf16 xf[b][i][d], set flags
__global__ __launch_bounds__(256)
void gather_kernel(const float* __restrict__ x,
                   const int* __restrict__ sel,
                   unsigned short* __restrict__ xf,
                   int* __restrict__ flags) {
    const int chunks = DMODEL / 8;                         // 256 chunks of 8 elems
    int t = blockIdx.x * blockDim.x + threadIdx.x;         // grid exact: B*KPAD*chunks
    int c = t & (chunks - 1);
    int i = (t / chunks) & (KPAD - 1);
    int b = t / (chunks * KPAD);
    int idx = sel[b * KPAD + i];
    uint4 outv = make_uint4(0u, 0u, 0u, 0u);
    if (idx >= 0) {
        const float4* src = (const float4*)(x + ((size_t)b * SEQ + idx) * DMODEL + c * 8);
        float4 lo = src[0], hi = src[1];
        outv.x = pack2bf(lo.x, lo.y);
        outv.y = pack2bf(lo.z, lo.w);
        outv.z = pack2bf(hi.x, hi.y);
        outv.w = pack2bf(hi.z, hi.w);
        if (c == 0) flags[b * SEQ + idx] = 1;
    }
    *(uint4*)(xf + ((size_t)b * KPAD + i) * DMODEL + c * 8) = outv;
}

// =====================================================================
// GEMM tiling
// =====================================================================
#define BM 128
#define BN 64
#define BK 64
#define LDK 72      // padded K stride (halfwords) to break LDS bank conflicts

// load one lane's 16x16x32 bf16 fragment (ISA 16-bit A/B VGPR layout):
// element dword v (0..7) holds K = 16*(v/4) + 8*(lane/16) + 2*(v%4) .. +1
__device__ __forceinline__ void load_frag(Frag16& f, const unsigned short* base,
                                          int rowcol, int ks, int half) {
#pragma unroll
    for (int v = 0; v < 8; ++v) {
        int kIdx = ks + ((v >> 2) << 4) + (half << 3) + ((v & 3) << 1);
        f.u[v] = *(const unsigned*)&base[rowcol * LDK + kIdx];
    }
}

// =====================================================================
// 4) GEMM1 fused: h = silu(xf@W1) * (xf@W3), bf16 out. A=[KPAD,D], B=[D,H]
// =====================================================================
__global__ __launch_bounds__(256)
void gemm1_kernel(const unsigned short* __restrict__ xf,
                  const float* __restrict__ W1,
                  const float* __restrict__ W3,
                  unsigned short* __restrict__ hact) {
    __shared__ unsigned short As [BM * LDK];
    __shared__ unsigned short B1s[BN * LDK];
    __shared__ unsigned short B3s[BN * LDK];

    const int b  = blockIdx.z;
    const int m0 = blockIdx.y * BM;
    const int n0 = blockIdx.x * BN;
    const int tid  = threadIdx.x;
    const int wave = tid >> 5, lane = tid & 31;
    const int waveM = wave >> 1, waveN = wave & 1;       // 4x2 wave grid
    const int half = lane >> 4, l16 = lane & 15;

    const unsigned* Ag = (const unsigned*)(xf + ((size_t)b * KPAD + m0) * DMODEL);

    f32x8 acc1[2][2] = {};
    f32x8 acc3[2][2] = {};

    for (int k0 = 0; k0 < DMODEL; k0 += BK) {
        // stage A tile: 128 rows x 32 dwords (2 bf16 each)
#pragma unroll
        for (int r = 0; r < 16; ++r) {
            int linear = tid + r * 256;
            int row = linear >> 5, kc = linear & 31;
            unsigned v = Ag[(size_t)row * (DMODEL / 2) + (k0 >> 1) + kc];
            *(unsigned*)&As[row * LDK + kc * 2] = v;
        }
        // stage B tiles (transpose to [n][k], fp32 -> bf16)
#pragma unroll
        for (int r = 0; r < 16; ++r) {
            int linear = tid + r * 256;
            int kr = linear >> 6, nc = linear & 63;
            size_t g = (size_t)(k0 + kr) * HID + (n0 + nc);
            B1s[nc * LDK + kr] = f2bf(W1[g]);
            B3s[nc * LDK + kr] = f2bf(W3[g]);
        }
        if (k0 + BK < DMODEL) {
            __builtin_prefetch(&W1[(size_t)(k0 + BK + (tid >> 6)) * HID + n0 + (tid & 63)], 0, 1);
            __builtin_prefetch(&W3[(size_t)(k0 + BK + (tid >> 6)) * HID + n0 + (tid & 63)], 0, 1);
        }
        __syncthreads();

#pragma unroll
        for (int ks = 0; ks < BK; ks += 32) {
            Frag16 a[2], b1[2], b3[2];
#pragma unroll
            for (int sm = 0; sm < 2; ++sm)
                load_frag(a[sm], As, waveM * 32 + sm * 16 + l16, ks, half);
#pragma unroll
            for (int sn = 0; sn < 2; ++sn) {
                load_frag(b1[sn], B1s, waveN * 32 + sn * 16 + l16, ks, half);
                load_frag(b3[sn], B3s, waveN * 32 + sn * 16 + l16, ks, half);
            }
#pragma unroll
            for (int sm = 0; sm < 2; ++sm)
#pragma unroll
                for (int sn = 0; sn < 2; ++sn) {
                    acc1[sm][sn] = __builtin_amdgcn_wmma_f32_16x16x32_bf16(
                        false, a[sm].v, false, b1[sn].v, (short)0, acc1[sm][sn], false, false);
                    acc3[sm][sn] = __builtin_amdgcn_wmma_f32_16x16x32_bf16(
                        false, a[sm].v, false, b3[sn].v, (short)0, acc3[sm][sn], false, false);
                }
        }
        __syncthreads();
    }

    // epilogue: silu(gate) * up -> bf16
#pragma unroll
    for (int sm = 0; sm < 2; ++sm)
#pragma unroll
        for (int sn = 0; sn < 2; ++sn) {
            int col = n0 + waveN * 32 + sn * 16 + l16;
#pragma unroll
            for (int r = 0; r < 8; ++r) {
                int row = m0 + waveM * 32 + sm * 16 + half * 8 + r;
                float g = acc1[sm][sn][r];
                float s = g / (1.f + __expf(-g));
                hact[((size_t)b * KPAD + row) * HID + col] = f2bf(s * acc3[sm][sn][r]);
            }
        }
}

// =====================================================================
// 5) GEMM2 + scatter: out[b, sel[i], :] += rw[i] * (h @ W2)
// =====================================================================
__global__ __launch_bounds__(256)
void gemm2_kernel(const unsigned short* __restrict__ hact,
                  const float* __restrict__ W2,
                  const int* __restrict__ sel,
                  const float* __restrict__ rw,
                  float* __restrict__ out) {
    __shared__ unsigned short As[BM * LDK];
    __shared__ unsigned short Bs[BN * LDK];

    const int b  = blockIdx.z;
    const int m0 = blockIdx.y * BM;
    const int n0 = blockIdx.x * BN;
    const int tid  = threadIdx.x;
    const int wave = tid >> 5, lane = tid & 31;
    const int waveM = wave >> 1, waveN = wave & 1;
    const int half = lane >> 4, l16 = lane & 15;

    const unsigned* Ag = (const unsigned*)(hact + ((size_t)b * KPAD + m0) * HID);

    f32x8 acc[2][2] = {};

    for (int k0 = 0; k0 < HID; k0 += BK) {
#pragma unroll
        for (int r = 0; r < 16; ++r) {
            int linear = tid + r * 256;
            int row = linear >> 5, kc = linear & 31;
            unsigned v = Ag[(size_t)row * (HID / 2) + (k0 >> 1) + kc];
            *(unsigned*)&As[row * LDK + kc * 2] = v;
        }
#pragma unroll
        for (int r = 0; r < 16; ++r) {
            int linear = tid + r * 256;
            int kr = linear >> 6, nc = linear & 63;
            Bs[nc * LDK + kr] = f2bf(W2[(size_t)(k0 + kr) * DMODEL + (n0 + nc)]);
        }
        if (k0 + BK < HID)
            __builtin_prefetch(&W2[(size_t)(k0 + BK + (tid >> 6)) * DMODEL + n0 + (tid & 63)], 0, 1);
        __syncthreads();

#pragma unroll
        for (int ks = 0; ks < BK; ks += 32) {
            Frag16 a[2], bb[2];
#pragma unroll
            for (int sm = 0; sm < 2; ++sm)
                load_frag(a[sm], As, waveM * 32 + sm * 16 + l16, ks, half);
#pragma unroll
            for (int sn = 0; sn < 2; ++sn)
                load_frag(bb[sn], Bs, waveN * 32 + sn * 16 + l16, ks, half);
#pragma unroll
            for (int sm = 0; sm < 2; ++sm)
#pragma unroll
                for (int sn = 0; sn < 2; ++sn)
                    acc[sm][sn] = __builtin_amdgcn_wmma_f32_16x16x32_bf16(
                        false, a[sm].v, false, bb[sn].v, (short)0, acc[sm][sn], false, false);
        }
        __syncthreads();
    }

    // scatter-add epilogue (unique tokens per batch -> plain RMW)
#pragma unroll
    for (int sm = 0; sm < 2; ++sm)
#pragma unroll
        for (int sn = 0; sn < 2; ++sn) {
            int col = n0 + waveN * 32 + sn * 16 + l16;
#pragma unroll
            for (int r = 0; r < 8; ++r) {
                int row = m0 + waveM * 32 + sm * 16 + half * 8 + r;
                if (row < KSEL) {
                    int   token = sel[b * KPAD + row];
                    float w     = rw [b * KPAD + row];
                    size_t o = ((size_t)b * SEQ + token) * DMODEL + col;
                    out[o] += w * acc[sm][sn][r];
                }
            }
        }
}

// =====================================================================
// 6) aux BCE reduction
// =====================================================================
__global__ __launch_bounds__(256)
void aux_kernel(const float* __restrict__ auxlog,
                const int* __restrict__ flags,
                float* __restrict__ aux_slot) {
    __shared__ float red[256];
    int n = blockIdx.x * 256 + threadIdx.x;              // grid exact: B*S
    float z = auxlog[n];
    float p = 1.f / (1.f + __expf(-z));
    const float eps = 1e-7f;
    p = fminf(fmaxf(p, eps), 1.f - eps);
    float tgt = flags[n] ? 1.f : 0.f;
    float term = -(tgt * __logf(p) + (1.f - tgt) * log1pf(-p)) * (1.f / (float)(BATCH * SEQ));
    red[threadIdx.x] = term;
    __syncthreads();
    for (int s = 128; s > 0; s >>= 1) {
        if (threadIdx.x < s) red[threadIdx.x] += red[threadIdx.x + s];
        __syncthreads();
    }
    if (threadIdx.x == 0) atomicAdd(aux_slot, red[0]);
}

// =====================================================================
// launcher
// =====================================================================
extern "C" void kernel_launch(void* const* d_in, const int* in_sizes, int n_in,
                              void* d_out, int out_size, void* d_ws, size_t ws_size,
                              hipStream_t stream) {
    (void)in_sizes; (void)n_in; (void)out_size; (void)ws_size;
    const float* x  = (const float*)d_in[0];
    const float* Wr = (const float*)d_in[1];
    const float* Wa = (const float*)d_in[2];
    const float* W1 = (const float*)d_in[3];
    const float* W2 = (const float*)d_in[4];   // note dict order: W2 before W3
    const float* W3 = (const float*)d_in[5];

    float* out = (float*)d_out;
    float* aux_slot = out + (size_t)BATCH * SEQ * DMODEL;

    // workspace carve-up
    char* ws = (char*)d_ws;
    float*          logits = (float*)(ws);                                   // 64 KB
    float*          auxlog = (float*)(ws + (1u << 16));                      // 64 KB
    int*            sel    = (int*)  (ws + (2u << 16));                      // 8 KB
    float*          rw     = (float*)(ws + (2u << 16) + 8192);               // 8 KB
    int*            flags  = (int*)  (ws + (2u << 16) + 16384);              // 64 KB
    unsigned short* xf     = (unsigned short*)(ws + (3u << 16) + 16384);     // 8 MB
    unsigned short* hact   = (unsigned short*)(ws + (3u << 16) + 16384
                                               + (size_t)BATCH * KPAD * DMODEL * 2); // 32 MB

    // residual copy x -> out (device-to-device, capture-safe)
    hipMemcpyAsync(out, x, (size_t)BATCH * SEQ * DMODEL * sizeof(float),
                   hipMemcpyDeviceToDevice, stream);

    router_kernel<<<(BATCH * SEQ) / 8, 256, 0, stream>>>(x, Wr, Wa, logits, auxlog, aux_slot);
    topk_kernel<<<BATCH, 512, 0, stream>>>(logits, sel, rw);
    zero_flags_kernel<<<(BATCH * SEQ) / 256, 256, 0, stream>>>(flags);
    gather_kernel<<<(BATCH * KPAD * (DMODEL / 8)) / 256, 256, 0, stream>>>(x, sel, xf, flags);

    gemm1_kernel<<<dim3(HID / BN, KPAD / BM, BATCH), 256, 0, stream>>>(xf, W1, W3, hact);
    gemm2_kernel<<<dim3(DMODEL / BN, KPAD / BM, BATCH), 256, 0, stream>>>(hact, W2, sel, rw, out);

    aux_kernel<<<(BATCH * SEQ) / 256, 256, 0, stream>>>(auxlog, flags, aux_slot);
}
